// CfCCell_5669356836202
// MI455X (gfx1250) — matrix operand order repaired and checked by reading, not compiled, not run on hardware
//
#include <hip/hip_runtime.h>
#include <hip/hip_bf16.h>
#include <stdint.h>

// ---------------- problem constants ----------------
#define B_ROWS  16384
#define IN_DIM  256
#define HID_DIM 512
#define CAT_DIM 768      // IN + HID
#define UNITS   1024
#define LDA_PAD 72       // 64 + 8 pad (bank spread, keeps 16B alignment)

// ---------------- WMMA types ----------------
typedef __attribute__((ext_vector_type(16))) __bf16 bf16x16;
typedef __attribute__((ext_vector_type(8)))  __bf16 bf16x8;
typedef __attribute__((ext_vector_type(8)))  float  f32x8;

union Frag16 { bf16x16 v; bf16x8 h[2]; };

__device__ __forceinline__ unsigned short f32_to_bf16(float f) {
  unsigned int u = __float_as_uint(f);
  u += 0x7fffu + ((u >> 16) & 1u);   // round-to-nearest-even
  return (unsigned short)(u >> 16);
}

__device__ __forceinline__ float fast_tanh(float x) {
  x = fminf(fmaxf(x, -15.f), 15.f);
  float e = __expf(2.f * x);
  return (e - 1.f) / (e + 1.f);
}

__device__ __forceinline__ float fast_sigmoid(float x) {
  return 1.f / (1.f + __expf(-x));
}

__device__ __forceinline__ f32x8 wmma_bf16(bf16x16 a, bf16x16 b, f32x8 c) {
  // (neg_a, A, neg_b, B, c_mod, C, reuse_a, reuse_b)
  return __builtin_amdgcn_wmma_f32_16x16x32_bf16(false, a, false, b, (short)0, c,
                                                 false, false);
}

// ---------------- gfx1250 async global->LDS copy (ASYNCcnt) ----------------
// Copies 64B per lane: IOFFSET is applied to BOTH the LDS and global addresses
// (ISA 08_async_tensor §4.4), so one LDS-addr VGPR + one global-addr VGPR stage
// a contiguous 64B chunk with offset:0/16/32/48.
__device__ __forceinline__ void async_copy_64B(unsigned int lds_off, const void* g) {
  asm volatile(
      "global_load_async_to_lds_b128 %0, %1, off\n\t"
      "global_load_async_to_lds_b128 %0, %1, off offset:16\n\t"
      "global_load_async_to_lds_b128 %0, %1, off offset:32\n\t"
      "global_load_async_to_lds_b128 %0, %1, off offset:48"
      :: "v"(lds_off), "v"(g) : "memory");
}

__device__ __forceinline__ void wait_async0() {
  asm volatile("s_wait_asynccnt 0x0" ::: "memory");
}

// Stage a 128x64 bf16 A tile into LDS via async DMA (256 threads, 64B each).
__device__ __forceinline__ void async_stage_tile(unsigned short* ldsA,
                                                 const unsigned short* __restrict__ src,
                                                 int m0, int k0, int K, int tid) {
  int row = tid >> 1;              // 0..127
  int seg = (tid & 1) * 32;        // 0 or 32 elements (64B half-row)
  const unsigned short* g = src + (size_t)(m0 + row) * K + k0 + seg;
  unsigned int lds_off = (unsigned int)(uintptr_t)(ldsA + row * LDA_PAD + seg);
  async_copy_64B(lds_off, g);
}

// A fragment (16x32 bf16) from LDS tile, row-major, stride LDA_PAD elements.
// Contiguous K order (same as B): lanes 0-15 hold M=lane, K=0..15; lanes 16-31 K=16..31.
__device__ __forceinline__ bf16x16 lds_a_frag(const unsigned short* lds, int lane,
                                              int mtile, int koff) {
  int row = mtile * 16 + (lane & 15);
  int kb  = koff + ((lane >> 4) * 16);
  Frag16 f;
  f.h[0] = *(const bf16x8*)(lds + row * LDA_PAD + kb);
  f.h[1] = *(const bf16x8*)(lds + row * LDA_PAD + kb + 8);
  return f.v;
}

// B fragment (32x16 bf16) straight from global bf16 weights stored [N][K] row-major
// (torch [out,in] layout). Lane holds column N=lane%16, 16 contiguous K values.
__device__ __forceinline__ bf16x16 glb_b_frag(const unsigned short* __restrict__ w,
                                              int lane, int nbase, int k0, int ldk) {
  int col = nbase + (lane & 15);
  int kb  = k0 + (lane >> 4) * 16;
  const unsigned short* p = w + (size_t)col * ldk + kb;
  Frag16 f;
  f.h[0] = *(const bf16x8*)(p);
  f.h[1] = *(const bf16x8*)(p + 8);
  return f.v;
}

// ---------------- prologue kernels ----------------
__global__ __launch_bounds__(256) void pack_x_kernel(
    const float* __restrict__ in, const float* __restrict__ hx,
    unsigned short* __restrict__ xb) {
  int idx = blockIdx.x * 256 + threadIdx.x;     // over B_ROWS*CAT_DIM
  if (idx >= B_ROWS * CAT_DIM) return;
  int b = idx / CAT_DIM, k = idx - b * CAT_DIM;
  float v = (k < IN_DIM) ? in[(size_t)b * IN_DIM + k]
                         : hx[(size_t)b * HID_DIM + (k - IN_DIM)];
  xb[idx] = f32_to_bf16(v);
}

__global__ __launch_bounds__(256) void cvt_bf16_kernel(
    const float* __restrict__ src, unsigned short* __restrict__ dst, int n) {
  int i = blockIdx.x * 256 + threadIdx.x;
  if (i < n) dst[i] = f32_to_bf16(src[i]);
}

// ---------------- backbone inner sub-step: 8 WMMAs over one 32-wide K slice ----
__device__ __forceinline__ void bb_substep(const unsigned short* cur,
                                           const unsigned short* __restrict__ wb,
                                           int lane, int waveM, int nwave, int kk,
                                           int koff, f32x8 acc[4][2]) {
  bf16x16 afr[4];
#pragma unroll
  for (int mi = 0; mi < 4; ++mi)
    afr[mi] = lds_a_frag(cur, lane, waveM * 4 + mi, koff);

  bf16x16 bfr[2];
#pragma unroll
  for (int ni = 0; ni < 2; ++ni)
    bfr[ni] = glb_b_frag(wb, lane, nwave + ni * 16, kk, CAT_DIM);

#pragma unroll
  for (int mi = 0; mi < 4; ++mi)
#pragma unroll
    for (int ni = 0; ni < 2; ++ni)
      acc[mi][ni] = wmma_bf16(afr[mi], bfr[ni], acc[mi][ni]);
}

// ---------------- backbone: h = lecun_tanh(x @ Wb^T + bb) ----------------
// Block = 128(M) x 128(N) tile, 8 waves in 2(M) x 4(N) grid, 64x32 per wave.
// Async-DMA double-buffered LDS A tile, K-step 64 (16 WMMA per barrier).
__global__ __launch_bounds__(256) void backbone_kernel(
    const unsigned short* __restrict__ xb,   // [B_ROWS][CAT_DIM] bf16
    const unsigned short* __restrict__ wb,   // [UNITS][CAT_DIM]  bf16
    const float* __restrict__ bb,            // [UNITS]
    unsigned short* __restrict__ hout) {     // [B_ROWS][UNITS]   bf16
  __shared__ unsigned short ldsA[2][128 * LDA_PAD];

  int tid  = threadIdx.x;
  int lane = tid & 31;
  int wave = tid >> 5;
  int waveM = wave & 1;                      // 2 waves along M (64 rows each)
  int waveN = wave >> 1;                     // 4 waves along N (32 cols each)
  int m0 = blockIdx.y * 128;
  int n0 = blockIdx.x * 128;
  int nwave = n0 + waveN * 32;

  f32x8 acc[4][2];
#pragma unroll
  for (int mi = 0; mi < 4; ++mi)
#pragma unroll
    for (int ni = 0; ni < 2; ++ni) acc[mi][ni] = (f32x8){};

  async_stage_tile(ldsA[0], xb, m0, 0, CAT_DIM, tid);
  wait_async0();
  __syncthreads();

  int buf = 0;
  for (int k0 = 0; k0 < CAT_DIM; k0 += 64) {
    bool more = (k0 + 64) < CAT_DIM;
    const unsigned short* cur = ldsA[buf];

    if (more) {
      // DMA next A tile into the alternate buffer; no VGPRs involved.
      async_stage_tile(ldsA[buf ^ 1], xb, m0, k0 + 64, CAT_DIM, tid);
      // pull next weight K-slice toward the WGP (global_prefetch_b8)
      __builtin_prefetch(wb + (size_t)(nwave + (lane & 15)) * CAT_DIM + k0 + 64, 0, 3);
    }

    // 16 WMMAs out of the current buffer cover the DMA latency.
    bb_substep(cur, wb, lane, waveM, nwave, k0, 0, acc);
    bb_substep(cur, wb, lane, waveM, nwave, k0 + 32, 32, acc);

    if (more) wait_async0();
    __syncthreads();
    buf ^= 1;
  }

  // Epilogue: bias + 1.7159*tanh(0.666*x), store bf16 h.
  int mlane = (lane >> 4) * 8;   // C/D layout: lanes>=16 hold M+8
  int ncol  = lane & 15;
#pragma unroll
  for (int ni = 0; ni < 2; ++ni) {
    int col = nwave + ni * 16 + ncol;
    float bias = bb[col];
#pragma unroll
    for (int mi = 0; mi < 4; ++mi) {
#pragma unroll
      for (int r = 0; r < 8; ++r) {
        int row = m0 + waveM * 64 + mi * 16 + mlane + r;
        float v = acc[mi][ni][r] + bias;
        v = 1.7159f * fast_tanh(0.666f * v);
        hout[(size_t)row * UNITS + col] = f32_to_bf16(v);
      }
    }
  }
}

// ---------------- heads inner sub-step: 8 WMMAs over one 32-wide K slice ------
__device__ __forceinline__ void hd_substep(const unsigned short* cur,
                                           const unsigned short* __restrict__ w1,
                                           const unsigned short* __restrict__ w2,
                                           const unsigned short* __restrict__ wa,
                                           const unsigned short* __restrict__ wt,
                                           int lane, int waveM, int nbase, int kk,
                                           int koff, f32x8 acc[2][4]) {
  bf16x16 afr[2];
#pragma unroll
  for (int mi = 0; mi < 2; ++mi)
    afr[mi] = lds_a_frag(cur, lane, waveM * 2 + mi, koff);

  bf16x16 bf1 = glb_b_frag(w1, lane, nbase, kk, UNITS);
  bf16x16 bf2 = glb_b_frag(w2, lane, nbase, kk, UNITS);
  bf16x16 bfa = glb_b_frag(wa, lane, nbase, kk, UNITS);
  bf16x16 bft = glb_b_frag(wt, lane, nbase, kk, UNITS);

#pragma unroll
  for (int mi = 0; mi < 2; ++mi) {
    acc[mi][0] = wmma_bf16(afr[mi], bf1, acc[mi][0]);
    acc[mi][1] = wmma_bf16(afr[mi], bf2, acc[mi][1]);
    acc[mi][2] = wmma_bf16(afr[mi], bfa, acc[mi][2]);
    acc[mi][3] = wmma_bf16(afr[mi], bft, acc[mi][3]);
  }
}

// ---------------- heads: fused 4x GEMM + gating ----------------
// Block = 128(M) x 32(N) tile, 8 waves in 4(M) x 2(N) grid, 32x16 per wave.
// Each wave keeps all four head accumulators for its tile -> zero-traffic epilogue.
__global__ __launch_bounds__(256) void heads_kernel(
    const unsigned short* __restrict__ h,    // [B_ROWS][UNITS] bf16
    const unsigned short* __restrict__ w1,   // [HID][UNITS] bf16
    const unsigned short* __restrict__ w2,
    const unsigned short* __restrict__ wa,
    const unsigned short* __restrict__ wt,
    const float* __restrict__ b1, const float* __restrict__ b2,
    const float* __restrict__ ba, const float* __restrict__ bt,
    const float* __restrict__ ts,            // [B_ROWS]
    float* __restrict__ out) {               // [B_ROWS][HID]
  __shared__ unsigned short ldsA[2][128 * LDA_PAD];

  int tid  = threadIdx.x;
  int lane = tid & 31;
  int wave = tid >> 5;
  int waveM = wave & 3;                      // 4 waves along M (32 rows each)
  int waveN = wave >> 2;                     // 2 waves along N (16 cols each)
  int m0 = blockIdx.y * 128;
  int n0 = blockIdx.x * 32;
  int nbase = n0 + waveN * 16;

  f32x8 acc[2][4];                           // [m subtile][head]
#pragma unroll
  for (int mi = 0; mi < 2; ++mi)
#pragma unroll
    for (int hd = 0; hd < 4; ++hd) acc[mi][hd] = (f32x8){};

  async_stage_tile(ldsA[0], h, m0, 0, UNITS, tid);
  wait_async0();
  __syncthreads();

  int buf = 0;
  for (int k0 = 0; k0 < UNITS; k0 += 64) {
    bool more = (k0 + 64) < UNITS;
    const unsigned short* cur = ldsA[buf];

    if (more) {
      async_stage_tile(ldsA[buf ^ 1], h, m0, k0 + 64, UNITS, tid);
      __builtin_prefetch(w1 + (size_t)(nbase + (lane & 15)) * UNITS + k0 + 64, 0, 3);
    }

    hd_substep(cur, w1, w2, wa, wt, lane, waveM, nbase, k0, 0, acc);
    hd_substep(cur, w1, w2, wa, wt, lane, waveM, nbase, k0 + 32, 32, acc);

    if (more) wait_async0();
    __syncthreads();
    buf ^= 1;
  }

  // Fused gating epilogue.
  int mlane = (lane >> 4) * 8;
  int col   = nbase + (lane & 15);
  float bias1 = b1[col], bias2 = b2[col], biasa = ba[col], biast = bt[col];
#pragma unroll
  for (int mi = 0; mi < 2; ++mi) {
#pragma unroll
    for (int r = 0; r < 8; ++r) {
      int row = m0 + waveM * 32 + mi * 16 + mlane + r;
      float tsv = ts[row];
      float f1 = fast_tanh(acc[mi][0][r] + bias1);
      float f2 = fast_tanh(acc[mi][1][r] + bias2);
      float ta = acc[mi][2][r] + biasa;
      float tb = acc[mi][3][r] + biast;
      float ti = fast_sigmoid(ta * tsv + tb);
      out[(size_t)row * HID_DIM + col] = f1 * (1.f - ti) + ti * f2;
    }
  }
}

// ---------------- workspace layout (bf16 elements) ----------------
static const size_t XB_OFF = 0;                                            // B*768
static const size_t WB_OFF = XB_OFF + (size_t)B_ROWS * CAT_DIM;            // 1024*768
static const size_t W1_OFF = WB_OFF + (size_t)UNITS * CAT_DIM;             // 512*1024
static const size_t W2_OFF = W1_OFF + (size_t)HID_DIM * UNITS;
static const size_t WA_OFF = W2_OFF + (size_t)HID_DIM * UNITS;
static const size_t WT_OFF = WA_OFF + (size_t)HID_DIM * UNITS;
static const size_t H_OFF  = WT_OFF + (size_t)HID_DIM * UNITS;             // B*1024

extern "C" void kernel_launch(void* const* d_in, const int* in_sizes, int n_in,
                              void* d_out, int out_size, void* d_ws, size_t ws_size,
                              hipStream_t stream) {
  const float* input = (const float*)d_in[0];
  const float* hx    = (const float*)d_in[1];
  const float* ts    = (const float*)d_in[2];
  const float* Wb    = (const float*)d_in[3];
  const float* bb    = (const float*)d_in[4];
  const float* W1    = (const float*)d_in[5];
  const float* b1    = (const float*)d_in[6];
  const float* W2    = (const float*)d_in[7];
  const float* b2    = (const float*)d_in[8];
  const float* Wa    = (const float*)d_in[9];
  const float* ba    = (const float*)d_in[10];
  const float* Wt    = (const float*)d_in[11];
  const float* bt    = (const float*)d_in[12];
  float* out = (float*)d_out;
  unsigned short* ws = (unsigned short*)d_ws;

  // 1) pack x = [input | hx] -> bf16
  {
    int n = B_ROWS * CAT_DIM;
    pack_x_kernel<<<(n + 255) / 256, 256, 0, stream>>>(input, hx, ws + XB_OFF);
  }
  // 2) weights fp32 -> bf16 (keep [out,in] layout == WMMA B-operand layout)
  {
    int nb = UNITS * CAT_DIM;
    cvt_bf16_kernel<<<(nb + 255) / 256, 256, 0, stream>>>(Wb, ws + WB_OFF, nb);
    int nh = HID_DIM * UNITS;
    cvt_bf16_kernel<<<(nh + 255) / 256, 256, 0, stream>>>(W1, ws + W1_OFF, nh);
    cvt_bf16_kernel<<<(nh + 255) / 256, 256, 0, stream>>>(W2, ws + W2_OFF, nh);
    cvt_bf16_kernel<<<(nh + 255) / 256, 256, 0, stream>>>(Wa, ws + WA_OFF, nh);
    cvt_bf16_kernel<<<(nh + 255) / 256, 256, 0, stream>>>(Wt, ws + WT_OFF, nh);
  }
  // 3) backbone GEMM + lecun_tanh -> h (bf16)
  {
    dim3 grid(UNITS / 128, B_ROWS / 128);    // (8, 128)
    backbone_kernel<<<grid, 256, 0, stream>>>(ws + XB_OFF, ws + WB_OFF, bb,
                                              ws + H_OFF);
  }
  // 4) four fused head GEMMs + gating -> out (fp32)
  {
    dim3 grid(HID_DIM / 32, B_ROWS / 128);   // (16, 128)
    heads_kernel<<<grid, 256, 0, stream>>>(ws + H_OFF,
                                           ws + W1_OFF, ws + W2_OFF,
                                           ws + WA_OFF, ws + WT_OFF,
                                           b1, b2, ba, bt, ts, out);
  }
}
